// DQN_gat_41601053229967
// MI455X (gfx1250) — matrix operand order repaired and checked by reading, not compiled, not run on hardware
//
#include <hip/hip_runtime.h>
#include <hip/hip_bf16.h>

typedef __attribute__((ext_vector_type(16))) _Float16 v16h;
typedef __attribute__((ext_vector_type(8)))  float    v8f;

#define NEG_SLOPE 0.2f

// ---------------------------------------------------------------- utilities
__global__ void fill_f32(float* p, float v, long n) {
  long i = (long)blockIdx.x * blockDim.x + threadIdx.x;
  if (i < n) p[i] = v;
}
__global__ void fill_u32(unsigned* p, unsigned v, long n) {
  long i = (long)blockIdx.x * blockDim.x + threadIdx.x;
  if (i < n) p[i] = v;
}

// Pre-swizzle the [fin, fout] row-major f32 weight into zero-padded f16 tiles
// laid out as gemm_wmma reads its B operand (same k order as its A loads):
//   chunk index ((kt*NT + nt)*32 + lane) holds 16 contiguous halves, where
//   lanes 0-15 carry K = kt*32+0..15 and lanes 16-31 carry K = kt*32+16..31,
//   for column nt*16 + (lane&15).
__global__ void pad_w_swz(const float* __restrict__ W, _Float16* __restrict__ Bsw,
                          int fin, int fout, int Ktiles, int NT) {
  int idx = blockIdx.x * blockDim.x + threadIdx.x;
  int total = Ktiles * NT * 32 * 16;
  if (idx >= total) return;
  int i    = idx & 15;
  int lane = (idx >> 4) & 31;
  int rest = idx >> 9;
  int nt   = rest % NT;
  int kt   = rest / NT;
  int k   = kt * 32 + ((lane & 16) ? 16 : 0) + i;
  int col = nt * 16 + (lane & 15);
  float v = (k < fin && col < fout) ? W[(long)k * fout + col] : 0.0f;
  Bsw[idx] = (_Float16)v;
}

// ------------------------------------------------------------- WMMA GEMM
// One wave computes a full 16 x (NT*16) output row-block of C = A(f32)*B.
// A fragment is loaded/converted once per K-tile and reused by NT WMMAs.
// Grid: (M/16). M=100000 is an exact multiple of 16; K a multiple of 32;
// only uniform branches -> EXEC is all-ones around every WMMA.
template <int NT>
__global__ __launch_bounds__(32) void gemm_wmma(
    const float* __restrict__ A, int lda,
    const _Float16* __restrict__ Bsw,
    float* __restrict__ C, int ldc, int Ktiles) {
  const int  lane  = threadIdx.x;
  const int  l15   = lane & 15;
  const bool hi    = lane >= 16;
  const long mbase = (long)blockIdx.x * 16;

  v8f acc[NT];
#pragma unroll
  for (int t = 0; t < NT; ++t) acc[t] = (v8f){};

  // A 16x32 f16 layout: lanes0-15 halves0-15 = K0..15;
  //                     lanes16-31 halves0-15 = K16..31.
  const float* arow = A + (mbase + l15) * (long)lda + (hi ? 16 : 0);
  const v16h*  bsw  = (const v16h*)Bsw + lane;

  for (int kt = 0; kt < Ktiles; ++kt) {
    if (kt + 1 < Ktiles)                       // uniform condition
      __builtin_prefetch(arow + (kt + 1) * 32, 0, 1);
    v16h a;
    const float* ap = arow + kt * 32;
#pragma unroll
    for (int i = 0; i < 8; ++i) {
      a[i]     = (_Float16)ap[i];
      a[8 + i] = (_Float16)ap[8 + i];
    }
    const v16h* bk = bsw + (long)kt * NT * 32;
#pragma unroll
    for (int t = 0; t < NT; ++t) {
      v16h b = bk[t * 32];                     // 32B contiguous per lane
      acc[t] = __builtin_amdgcn_wmma_f32_16x16x32_f16(
          /*neg_a=*/false, a, /*neg_b=*/false, b,
          /*c_mod=*/(short)0, acc[t], /*reuse_a=*/false, /*reuse_b=*/false);
    }
  }
  // C layout: VGPR i -> row mbase + i + (hi?8:0), col t*16 + l15.
  float* crow = C + (mbase + (hi ? 8 : 0)) * (long)ldc + l15;
#pragma unroll
  for (int t = 0; t < NT; ++t)
#pragma unroll
    for (int i = 0; i < 8; ++i) crow[(long)i * ldc + t * 16] = acc[t][i];
}

// ----------------------------------------------------- attention logits
__global__ void attn_logits(const float* __restrict__ h, int ldh,
                            const float* __restrict__ a_src,
                            const float* __restrict__ a_dst,
                            float* __restrict__ es, float* __restrict__ ed,
                            long Nn, int heads, int C) {
  long idx = (long)blockIdx.x * blockDim.x + threadIdx.x;
  if (idx >= Nn * heads) return;
  long n = idx / heads;
  int  hd = (int)(idx % heads);
  const float* hp = h + n * (long)ldh + hd * C;
  const float* as = a_src + hd * C;
  const float* ad = a_dst + hd * C;
  float s = 0.f, d = 0.f;
  for (int c = 0; c < C; ++c) { s += hp[c] * as[c]; d += hp[c] * ad[c]; }
  es[idx] = s; ed[idx] = d;
}

// --------------------------------------------------------- edge kernels
__device__ inline void edge_endpoints(const int* ei, long E, long e, int& s, int& d) {
  if (e < E) { s = ei[e]; d = ei[E + e]; }
  else       { s = d = (int)(e - E); }       // appended self loops
}
__device__ inline unsigned f2ord(float f) {
  unsigned u = __float_as_uint(f);
  return (u & 0x80000000u) ? ~u : (u | 0x80000000u);
}
__device__ inline float ord2f(unsigned u) {
  return (u & 0x80000000u) ? __uint_as_float(u ^ 0x80000000u) : __uint_as_float(~u);
}

__global__ void edge_max(const int* __restrict__ ei, long E, long ET, int heads,
                         const float* __restrict__ es, const float* __restrict__ ed,
                         float* __restrict__ ebuf, unsigned* __restrict__ mkeys) {
  long idx = (long)blockIdx.x * blockDim.x + threadIdx.x;
  if (idx >= ET * heads) return;
  long e = idx / heads; int hd = (int)(idx % heads);
  int s, d; edge_endpoints(ei, E, e, s, d);
  float v = es[(long)s * heads + hd] + ed[(long)d * heads + hd];
  v = v > 0.f ? v : NEG_SLOPE * v;             // leaky_relu
  ebuf[idx] = v;
  atomicMax(&mkeys[(long)d * heads + hd], f2ord(v));
}

__global__ void edge_exp(const int* __restrict__ ei, long E, long ET, int heads,
                         float* __restrict__ ebuf, const unsigned* __restrict__ mkeys,
                         float* __restrict__ z) {
  long idx = (long)blockIdx.x * blockDim.x + threadIdx.x;
  if (idx >= ET * heads) return;
  long e = idx / heads; int hd = (int)(idx % heads);
  int s, d; edge_endpoints(ei, E, e, s, d);
  float m = ord2f(mkeys[(long)d * heads + hd]);
  float w = __expf(ebuf[idx] - m);
  ebuf[idx] = w;
  atomicAdd(&z[(long)d * heads + hd], w);
}

__global__ void aggregate(const int* __restrict__ ei, long E, long ET, int heads, int C,
                          const float* __restrict__ ebuf, const float* __restrict__ z,
                          const float* __restrict__ h, int ldh,
                          float* __restrict__ out, int ldo) {
  long idx = (long)blockIdx.x * blockDim.x + threadIdx.x;
  if (idx >= ET * heads) return;
  long e = idx / heads; int hd = (int)(idx % heads);
  int s, d; edge_endpoints(ei, E, e, s, d);
  float alpha = ebuf[idx] / z[(long)d * heads + hd];
  const float* hp = h + (long)s * ldh + hd * C;
  float*       op = out + (long)d * ldo + hd * C;
  for (int c = 0; c < C; ++c) atomicAdd(&op[c], alpha * hp[c]);
}

__global__ void bias_act(float* __restrict__ out, int ldo, const float* __restrict__ b,
                         long Nn, int F, int relu) {
  long idx = (long)blockIdx.x * blockDim.x + threadIdx.x;
  if (idx >= Nn * F) return;
  long n = idx / F; int f = (int)(idx % F);
  float v = out[n * (long)ldo + f] + b[f];
  if (relu) v = fmaxf(v, 0.f);
  out[n * (long)ldo + f] = v;
}

__global__ void final_out(const float* __restrict__ agg, int ldo,
                          const float* __restrict__ b, const float* __restrict__ mask,
                          float* __restrict__ out, long Nn, int F) {
  long idx = (long)blockIdx.x * blockDim.x + threadIdx.x;
  if (idx >= Nn * F) return;
  long n = idx / F; int f = (int)(idx % F);
  out[idx] = agg[n * (long)ldo + f] + b[f] + (mask[idx] - 1.0f) * 1000.0f;
}

// ------------------------------------------------------------ launcher
extern "C" void kernel_launch(void* const* d_in, const int* in_sizes, int n_in,
                              void* d_out, int out_size, void* d_ws, size_t ws_size,
                              hipStream_t stream) {
  const int IN = 128, HID = 22, OUTF = 11, HEADS = 4;
  const long Nn = in_sizes[0] / IN;     // 100000 (multiple of 16)
  const long E  = in_sizes[1] / 2;      // 3,200,000
  const long ET = E + Nn;

  const float* x    = (const float*)d_in[0];
  const int*   ei   = (const int*)d_in[1];
  const float* mask = (const float*)d_in[2];
  const float *W[4], *As[4], *Ad[4], *Bb[4];
  for (int l = 0; l < 4; ++l) {
    W[l]  = (const float*)d_in[3 + 4 * l];
    As[l] = (const float*)d_in[4 + 4 * l];
    Ad[l] = (const float*)d_in[5 + 4 * l];
    Bb[l] = (const float*)d_in[6 + 4 * l];
  }

  // --- workspace carve-up (~137 MB); every block is 32B-aligned ---
  char* ws = (char*)d_ws;
  float* hg   = (float*)ws;  ws += Nn * 96 * sizeof(float);   // GEMM output
  float* agg  = (float*)ws;  ws += Nn * 96 * sizeof(float);   // aggregated output / next input
  float* esb  = (float*)ws;  ws += Nn * 4 * sizeof(float);
  float* edb  = (float*)ws;  ws += Nn * 4 * sizeof(float);
  unsigned* mk = (unsigned*)ws; ws += Nn * 4 * sizeof(unsigned);
  float* zb   = (float*)ws;  ws += Nn * 4 * sizeof(float);
  float* ebuf = (float*)ws;  ws += ET * 4 * sizeof(float);
  // swizzled f16 weights: Ktiles*NT*32 chunks of 16 halves
  _Float16* w1p = (_Float16*)ws; ws += (long)4 * 6 * 512 * sizeof(_Float16);
  _Float16* w2p = (_Float16*)ws; ws += (long)3 * 6 * 512 * sizeof(_Float16);
  _Float16* w3p = (_Float16*)ws; ws += (long)3 * 6 * 512 * sizeof(_Float16);
  _Float16* w4p = (_Float16*)ws; ws += (long)3 * 1 * 512 * sizeof(_Float16);

  const int TB = 256;
  // swizzled, zero-padded f16 weights
  pad_w_swz<<<(4 * 6 * 512 + TB - 1) / TB, TB, 0, stream>>>(W[0], w1p, 128, 88, 4, 6);
  pad_w_swz<<<(3 * 6 * 512 + TB - 1) / TB, TB, 0, stream>>>(W[1], w2p, 88, 88, 3, 6);
  pad_w_swz<<<(3 * 6 * 512 + TB - 1) / TB, TB, 0, stream>>>(W[2], w3p, 88, 88, 3, 6);
  pad_w_swz<<<(3 * 1 * 512 + TB - 1) / TB, TB, 0, stream>>>(W[3], w4p, 88, 11, 3, 1);

  auto layer = [&](const float* A, int lda, int Ktiles, const _Float16* Wp, int Npad,
                   int heads, int C, const float* a_s, const float* a_d,
                   const float* bias, int relu) {
    // h = A @ Wp  (into hg, leading dim Npad)
    if (Npad == 96)
      gemm_wmma<6><<<(unsigned)(Nn / 16), 32, 0, stream>>>(A, lda, Wp, hg, Npad, Ktiles);
    else
      gemm_wmma<1><<<(unsigned)(Nn / 16), 32, 0, stream>>>(A, lda, Wp, hg, Npad, Ktiles);

    long nh = Nn * heads;
    attn_logits<<<(nh + TB - 1) / TB, TB, 0, stream>>>(hg, Npad, a_s, a_d, esb, edb, Nn, heads, C);
    fill_u32<<<(nh + TB - 1) / TB, TB, 0, stream>>>(mk, 0u, nh);
    fill_f32<<<(nh + TB - 1) / TB, TB, 0, stream>>>(zb, 0.f, nh);

    long eh = ET * heads;
    edge_max<<<(eh + TB - 1) / TB, TB, 0, stream>>>(ei, E, ET, heads, esb, edb, ebuf, mk);
    edge_exp<<<(eh + TB - 1) / TB, TB, 0, stream>>>(ei, E, ET, heads, ebuf, mk, zb);

    long cnt = Nn * (long)Npad;  // zero full padded rows so pad columns stay 0
    fill_f32<<<(cnt + TB - 1) / TB, TB, 0, stream>>>(agg, 0.f, cnt);
    aggregate<<<(eh + TB - 1) / TB, TB, 0, stream>>>(ei, E, ET, heads, C, ebuf, zb, hg, Npad, agg, Npad);

    if (bias) {
      long nf = Nn * heads * C;
      bias_act<<<(nf + TB - 1) / TB, TB, 0, stream>>>(agg, Npad, bias, Nn, heads * C, relu);
    }
  };

  // Layer 1: X[100k,128] -> 4x22  (K=128 exact)
  layer(x, 128, 4, w1p, 96, HEADS, HID, As[0], Ad[0], Bb[0], 1);
  // Layers 2,3: 88 -> 88 (K padded to 96; agg serves as both input and, after
  // the GEMM has consumed it, the zeroed aggregation target — stream-ordered)
  layer(agg, 96, 3, w2p, 96, HEADS, HID, As[1], Ad[1], Bb[1], 1);
  layer(agg, 96, 3, w3p, 96, HEADS, HID, As[2], Ad[2], Bb[2], 1);
  // Layer 4: 88 -> 11 (Npad=16, heads=1); bias folded into final kernel
  layer(agg, 96, 3, w4p, 16, 1, OUTF, As[3], Ad[3], nullptr, 0);

  long tot = Nn * OUTF;
  final_out<<<(tot + TB - 1) / TB, TB, 0, stream>>>(agg, 16, Bb[3], mask, (float*)d_out, Nn, OUTF);
}
